// LightweightCCA_58239756534201
// MI455X (gfx1250) — matrix operand order repaired and checked by reading, not compiled, not run on hardware
//
#include <hip/hip_runtime.h>
#include <hip/hip_bf16.h>

// LightweightCCA for MI455X (gfx1250, wave32, WMMA).
// Pipeline: weightsT(bf16) -> QKV proj (bf16 WMMA) -> row attn -> col attn -> fused epilogue.
// v2: bf16 out_h/out_v buffers (no f32 RMW accumulator, saves ~268MB HBM traffic),
//     software double-buffering of B fragments and V tiles for load/WMMA overlap.

typedef __attribute__((ext_vector_type(16))) __bf16 v16bf;
typedef __attribute__((ext_vector_type(8)))  float  v8f;

#define Bsz 8
#define Cch 256
#define Hh  128
#define Ww  128
#define Dd  64

__device__ __forceinline__ v8f splat8(float v) {
    v8f r;
#pragma unroll
    for (int i = 0; i < 8; ++i) r[i] = v;
    return r;
}

__device__ __forceinline__ v8f wmma_bf16(v16bf a, v16bf b, v8f c) {
    // (neg_a, A, neg_b, B, c_mod, C, reuse_a, reuse_b)
    return __builtin_amdgcn_wmma_f32_16x16x32_bf16(false, a, false, b, (short)0, c, false, false);
}

// A fragment of a row-major [M x K] tile. k order shared with load_bT_frag, not the ISA table order (16-bit A 16x32):
// lane L: M = L&15, halves i -> K = kbase + 16*(L>>4) + i (contiguous)
__device__ __forceinline__ v16bf load_a_frag(const __bf16* A, int ldA, int kbase, int lane) {
    int m = lane & 15, hl = lane >> 4;
    const __bf16* p = A + (size_t)m * ldA + kbase + 16 * hl;
    v16bf a;
#pragma unroll
    for (int i = 0; i < 8; ++i) a[i] = p[i];
#pragma unroll
    for (int i = 0; i < 8; ++i) a[8 + i] = p[8 + i];
    return a;
}

// B fragment from a transposed buffer BT stored [N][K] row-major:
// lane L: N = L&15, halves i -> K = kbase + 16*(L>>4) + i (contiguous 32B per lane)
__device__ __forceinline__ v16bf load_bT_frag(const __bf16* BT, int ldBT, int kbase, int lane) {
    int n = lane & 15, hl = lane >> 4;
    const __bf16* p = BT + (size_t)n * ldBT + kbase + 16 * hl;
    v16bf b;
#pragma unroll
    for (int i = 0; i < 16; ++i) b[i] = p[i];
    return b;
}

// ---------- kernel 0: weight transpose + bf16 convert: src [R][Ccol] f32 -> dst [Ccol][R] bf16
__global__ void wt_kernel(const float* __restrict__ src, __bf16* __restrict__ dst, int R, int Ccol) {
    int i = blockIdx.x * 256 + threadIdx.x;
    if (i >= R * Ccol) return;
    int r = i / Ccol, c = i % Ccol;
    dst[(size_t)c * R + r] = (__bf16)src[i];
}

// ---------- kernel 1: QKV projection. One block per (b,h) row: 128 pixels, K=256.
__global__ __launch_bounds__(256) void qkv_kernel(
    const float* __restrict__ x,
    const __bf16* __restrict__ WqT, const float* __restrict__ bq,
    const __bf16* __restrict__ WkT, const float* __restrict__ bk,
    const __bf16* __restrict__ WvT, const float* __restrict__ bv,
    __bf16* __restrict__ qb, __bf16* __restrict__ kb, __bf16* __restrict__ vb) {
    extern __shared__ __bf16 xs[];      // [128][LDX] bf16 staging of the x row-slice
    const int LDX = 260;                // pad: 130 dwords -> conflict-free strided reads
    int bh = blockIdx.x;
    int b = bh >> 7, hrow = bh & 127;
    int t = threadIdx.x;
    const float* xbase = x + (((size_t)b * Cch) * Hh + hrow) * Ww;   // x[b][c][hrow][*]
    // coalesced load: each c-plane row of 128 floats -> bf16 LDS
#pragma unroll 4
    for (int it = 0; it < 128; ++it) {
        int c = it * 2 + (t >> 7);
        int w = t & 127;
        xs[w * LDX + c] = (__bf16)xbase[(size_t)c * (Hh * Ww) + w];
    }
    __syncthreads();

    int lane = t & 31, wv = t >> 5;
    int nl = lane & 15, hl = lane >> 4;
    int mbase = wv * 16;                                  // 8 waves x 16-pixel strips
    v16bf af[8];
#pragma unroll
    for (int ks = 0; ks < 8; ++ks)
        af[ks] = load_a_frag(xs + mbase * LDX, LDX, ks * 32, lane);

    size_t pixbase = (size_t)bh * Ww + mbase;             // flat pixel index of strip row 0

    // q (4 n-tiles), k (4), v (16); bias folded into accumulator init
#pragma unroll 1
    for (int seg = 0; seg < 3; ++seg) {
        const __bf16* WT = (seg == 0) ? WqT : (seg == 1) ? WkT : WvT;
        const float* bias = (seg == 0) ? bq : (seg == 1) ? bk : bv;
        __bf16* dst = (seg == 0) ? qb : (seg == 1) ? kb : vb;
        int Nt = (seg == 2) ? 16 : 4;
        int ldo = (seg == 2) ? Cch : Dd;
#pragma unroll 1
        for (int nt = 0; nt < Nt; ++nt) {
            const __bf16* Wrow = WT + (size_t)(nt * 16) * Cch;
            v8f acc = splat8(bias[nt * 16 + nl]);
            // double-buffered B fragments: keep one load in flight behind each WMMA
            v16bf b0 = load_bT_frag(Wrow, Cch, 0, lane);
#pragma unroll
            for (int ks = 0; ks < 8; ++ks) {
                v16bf b1 = (ks < 7) ? load_bT_frag(Wrow, Cch, (ks + 1) * 32, lane) : b0;
                acc = wmma_bf16(af[ks], b0, acc);
                b0 = b1;
            }
#pragma unroll
            for (int j = 0; j < 8; ++j) {
                int m = j + 8 * hl;                       // C/D layout: VGPR j -> M=j (+8 for hi half)
                dst[(pixbase + m) * ldo + nt * 16 + nl] = (__bf16)acc[j];
            }
        }
    }
}

// ---------- shared body for the two attention passes -------------------------
// q/k rows stride `qstride` halves (64 for row-attn, W*64 for col-attn), v rows `vstride`.
// writes bf16: outb[(pix0 + m*pixstep)*C + c]
__device__ __forceinline__ void attn_body(
    const __bf16* qstrip_base, const __bf16* kcol_base, const __bf16* v_base,
    int qstride, int vstride, __bf16* __restrict__ outb,
    size_t pix0, size_t pixstep,
    __bf16* attn_s, __bf16* vt) {
    int t = threadIdx.x, lane = t & 31, wv = t >> 5;
    int nl = lane & 15, hl = lane >> 4;
    int mbase = wv * 16;

    v16bf aq[2];
    aq[0] = load_a_frag(qstrip_base + (size_t)mbase * qstride, qstride, 0, lane);
    aq[1] = load_a_frag(qstrip_base + (size_t)mbase * qstride, qstride, 32, lane);

    // ---- scores = Q K^T, K fragments double-buffered across n-tiles
    float sc[8][8];
    v16bf kb0 = load_bT_frag(kcol_base, qstride, 0, lane);
    v16bf kb1 = load_bT_frag(kcol_base, qstride, 32, lane);
#pragma unroll 1
    for (int nt = 0; nt < 8; ++nt) {
        v16bf kn0 = kb0, kn1 = kb1;
        if (nt < 7) {
            kn0 = load_bT_frag(kcol_base + (size_t)((nt + 1) * 16) * qstride, qstride, 0, lane);
            kn1 = load_bT_frag(kcol_base + (size_t)((nt + 1) * 16) * qstride, qstride, 32, lane);
        }
        v8f acc = splat8(0.f);
        acc = wmma_bf16(aq[0], kb0, acc);
        acc = wmma_bf16(aq[1], kb1, acc);
#pragma unroll
        for (int j = 0; j < 8; ++j) sc[nt][j] = acc[j];
        kb0 = kn0; kb1 = kn1;
    }
    // ---- per-row softmax: row M=j+8*hl lives on 16 lanes of one half-wave across 8 n-tiles
#pragma unroll
    for (int j = 0; j < 8; ++j) {
        float mx = sc[0][j];
#pragma unroll
        for (int nt = 1; nt < 8; ++nt) mx = fmaxf(mx, sc[nt][j]);
#pragma unroll
        for (int d = 1; d < 16; d <<= 1) mx = fmaxf(mx, __shfl_xor(mx, d, 32));
        float s = 0.f;
#pragma unroll
        for (int nt = 0; nt < 8; ++nt) { float e = __expf(sc[nt][j] - mx); sc[nt][j] = e; s += e; }
#pragma unroll
        for (int d = 1; d < 16; d <<= 1) s += __shfl_xor(s, d, 32);
        float inv = 1.0f / s;
        int m = mbase + j + 8 * hl;
#pragma unroll
        for (int nt = 0; nt < 8; ++nt)
            attn_s[m * 130 + nt * 16 + nl] = (__bf16)(sc[nt][j] * inv);
    }

    // ---- out = P @ V: V tiles register-staged + LDS double-buffered (1 barrier/iter)
    __bf16 stg[8];
#pragma unroll
    for (int it = 0; it < 8; ++it) {
        int idx = it * 256 + t, u = idx >> 4, cl = idx & 15;
        stg[it] = v_base[(size_t)u * vstride + cl];
    }
#pragma unroll
    for (int it = 0; it < 8; ++it) {
        int idx = it * 256 + t, u = idx >> 4, cl = idx & 15;
        vt[cl * 130 + u] = stg[it];
    }
    __syncthreads();   // attn_s written by all waves + first V tile committed

#pragma unroll 1
    for (int ct = 0; ct < 16; ++ct) {
        __bf16* cur = vt + (ct & 1) * (16 * 130);
        __bf16* nxt = vt + ((ct + 1) & 1) * (16 * 130);
        if (ct + 1 < 16) {   // prefetch next V tile into registers (uniform branch)
#pragma unroll
            for (int it = 0; it < 8; ++it) {
                int idx = it * 256 + t, u = idx >> 4, cl = idx & 15;
                stg[it] = v_base[(size_t)u * vstride + (ct + 1) * 16 + cl];
            }
        }
        v8f acc = splat8(0.f);
#pragma unroll
        for (int ku = 0; ku < 4; ++ku) {
            v16bf a  = load_a_frag(attn_s + mbase * 130, 130, ku * 32, lane);
            v16bf bf = load_bT_frag(cur, 130, ku * 32, lane);
            acc = wmma_bf16(a, bf, acc);
        }
#pragma unroll
        for (int j = 0; j < 8; ++j) {
            int m = mbase + j + 8 * hl;
            size_t o = (pix0 + (size_t)m * pixstep) * Cch + ct * 16 + nl;
            outb[o] = (__bf16)acc[j];
        }
        if (ct + 1 < 16) {
#pragma unroll
            for (int it = 0; it < 8; ++it) {
                int idx = it * 256 + t, u = idx >> 4, cl = idx & 15;
                nxt[cl * 130 + u] = stg[it];
            }
        }
        __syncthreads();
    }
}

// ---------- kernel 2: horizontal (row) attention, one block per (b,h)
__global__ __launch_bounds__(256) void attn_h_kernel(
    const __bf16* __restrict__ qb, const __bf16* __restrict__ kb,
    const __bf16* __restrict__ vb, __bf16* __restrict__ ohb) {
    __shared__ __bf16 attn_s[128 * 130];
    __shared__ __bf16 vt[2 * 16 * 130];
    size_t rowpix = (size_t)blockIdx.x * Ww;      // pixel (b,h,0)
    attn_body(qb + rowpix * Dd, kb + rowpix * Dd, vb + rowpix * Cch,
              Dd, Cch, ohb, rowpix, 1, attn_s, vt);
}

// ---------- kernel 3: vertical (column) attention, one block per (b,w)
__global__ __launch_bounds__(256) void attn_v_kernel(
    const __bf16* __restrict__ qb, const __bf16* __restrict__ kb,
    const __bf16* __restrict__ vb, __bf16* __restrict__ ovb) {
    __shared__ __bf16 attn_s[128 * 130];
    __shared__ __bf16 vt[2 * 16 * 130];
    int bw = blockIdx.x;
    int b = bw >> 7, w = bw & 127;
    size_t colpix = (size_t)b * (Hh * Ww) + w;    // pixel (b,0,w); step = W per h
    attn_body(qb + colpix * Dd, kb + colpix * Dd, vb + colpix * Cch,
              Ww * Dd, Ww * Cch, ovb, colpix, Ww, attn_s, vt);
}

// ---------- kernel 4: out[b,c,h,w] = gamma*(oh+ov)[b,h,w,c] + x[b,c,h,w] (32x32 LDS transpose)
__global__ __launch_bounds__(256) void final_kernel(
    const float* __restrict__ x, const __bf16* __restrict__ oh, const __bf16* __restrict__ ov,
    const float* __restrict__ gamma, float* __restrict__ out) {
    __shared__ float tile[32 * 33];
    int blk = blockIdx.x;
    int wt = blk & 3, ct = (blk >> 2) & 7, bh = blk >> 5;
    int b = bh >> 7, h = bh & 127;
    int t = threadIdx.x;
    int c0 = ct * 32, w0 = wt * 32;
    float g = gamma[0];
#pragma unroll
    for (int i = 0; i < 4; ++i) {
        int cl = t & 31, wl = i * 8 + (t >> 5);
        size_t idx = ((size_t)bh * Ww + w0 + wl) * Cch + c0 + cl;
        tile[cl * 33 + wl] = (float)oh[idx] + (float)ov[idx];
    }
    __syncthreads();
#pragma unroll
    for (int i = 0; i < 4; ++i) {
        int wl = t & 31, cl = i * 8 + (t >> 5);
        size_t o = (((size_t)b * Cch + c0 + cl) * Hh + h) * Ww + w0 + wl;
        out[o] = g * tile[cl * 33 + wl] + x[o];
    }
}

extern "C" void kernel_launch(void* const* d_in, const int* in_sizes, int n_in,
                              void* d_out, int out_size, void* d_ws, size_t ws_size,
                              hipStream_t stream) {
    (void)in_sizes; (void)n_in; (void)out_size; (void)ws_size;
    const float* x     = (const float*)d_in[0];
    const float* Wq    = (const float*)d_in[1];
    const float* bq    = (const float*)d_in[2];
    const float* Wk    = (const float*)d_in[3];
    const float* bk    = (const float*)d_in[4];
    const float* Wv    = (const float*)d_in[5];
    const float* bv    = (const float*)d_in[6];
    const float* gamma = (const float*)d_in[7];
    float* out = (float*)d_out;

    // workspace carve-up (bytes): WqT 32K | WkT 32K | WvT 128K | q 16M | k 16M | v 64M | oh 64M | ov 64M
    char* ws = (char*)d_ws;
    __bf16* WqT = (__bf16*)ws;
    __bf16* WkT = (__bf16*)(ws + 32768);
    __bf16* WvT = (__bf16*)(ws + 65536);
    __bf16* qb  = (__bf16*)(ws + 196608);
    __bf16* kb  = qb + (size_t)Bsz * Hh * Ww * Dd;
    __bf16* vb  = kb + (size_t)Bsz * Hh * Ww * Dd;
    __bf16* ohb = vb + (size_t)Bsz * Hh * Ww * Cch;
    __bf16* ovb = ohb + (size_t)Bsz * Hh * Ww * Cch;

    wt_kernel<<<(Cch * Dd + 255) / 256, 256, 0, stream>>>(Wq, WqT, Cch, Dd);
    wt_kernel<<<(Cch * Dd + 255) / 256, 256, 0, stream>>>(Wk, WkT, Cch, Dd);
    wt_kernel<<<(Cch * Cch + 255) / 256, 256, 0, stream>>>(Wv, WvT, Cch, Cch);

    size_t qkv_lds = 128 * 260 * sizeof(__bf16);   // 66,560 B dynamic LDS (<= 320KB/WGP)
    qkv_kernel<<<Bsz * Hh, 256, qkv_lds, stream>>>(x, WqT, bq, WkT, bk, WvT, bv, qb, kb, vb);

    attn_h_kernel<<<Bsz * Hh, 256, 0, stream>>>(qb, kb, vb, ohb);
    attn_v_kernel<<<Bsz * Ww, 256, 0, stream>>>(qb, kb, vb, ovb);

    final_kernel<<<Bsz * Hh * 8 * 4, 256, 0, stream>>>(x, ohb, ovb, gamma, out);
}
